// RelativeMultiHeadAttention_7937099563319
// MI455X (gfx1250) — matrix operand order repaired and checked by reading, not compiled, not run on hardware
//
#include <hip/hip_runtime.h>

typedef __attribute__((ext_vector_type(16))) __bf16 v16bf;
typedef __attribute__((ext_vector_type(8)))  float  v8f;

constexpr int Bb = 2, Ss = 2048, Dd = 1024, Hh = 16, DHh = 64;
constexpr int NWAVE = 8;

// ---------- bf16 helpers ----------
__device__ __forceinline__ unsigned short f2bf(float f) {
  union { float f; unsigned u; } a; a.f = f;
  unsigned r = a.u + 0x7FFFu + ((a.u >> 16) & 1u);   // round-to-nearest-even
  return (unsigned short)(r >> 16);
}
__device__ __forceinline__ float bf2f(unsigned short h) {
  union { unsigned u; float f; } a; a.u = ((unsigned)h) << 16;
  return a.f;
}

union Frag { v16bf v; uint4 q[2]; };

// A-fragment (16x32 bf16, M x K): lane<16 -> row M=lane, K {0..7}U{16..23};
// lane>=16 -> row M=lane-16, K {8..15}U{24..31}. Two 16B contiguous chunks.
__device__ __forceinline__ v16bf load_afrag(const unsigned short* base, int stride,
                                            int row0, int col0, int lane) {
  int r = row0 + (lane & 15);
  int c = col0 + ((lane >> 4) << 3);
  const unsigned short* p = base + (size_t)r * stride + c;
  Frag f;
  f.q[0] = *(const uint4*)(p);
  f.q[1] = *(const uint4*)(p + 16);
  return f.v;
}
// B-fragment (32x16 bf16, K x N) read from "B-transposed" row-major source:
// B[kk][n] = src[n0+n][kk0+kk]. lane<16 -> col N=lane, K {0..7}U{16..23}; lane>=16 -> K {8..15}U{24..31}.
// Two 16B contiguous chunks per lane.
__device__ __forceinline__ v16bf load_bfrag(const unsigned short* src, int stride,
                                            int n0, int kk0, int lane) {
  int r = n0 + (lane & 15);
  int c = kk0 + ((lane >> 4) << 3);
  const unsigned short* p = src + (size_t)r * stride + c;
  Frag f;
  f.q[0] = *(const uint4*)(p);
  f.q[1] = *(const uint4*)(p + 16);
  return f.v;
}

#define WMMA_BF16(a, b, c) \
  __builtin_amdgcn_wmma_f32_16x16x32_bf16(false, (a), false, (b), (short)0, (c), false, false)

__device__ __forceinline__ v8f vzero8() {
  v8f z = {0.f, 0.f, 0.f, 0.f, 0.f, 0.f, 0.f, 0.f};
  return z;
}

// ---------- conversion kernels ----------
__global__ void cvt_bf16_kernel(const float* __restrict__ src,
                                unsigned short* __restrict__ dst, int n) {
  int i = blockIdx.x * blockDim.x + threadIdx.x;
  if (i < n) dst[i] = f2bf(src[i]);
}

__global__ void transpose_cvt_kernel(const float* __restrict__ W,
                                     unsigned short* __restrict__ Wt) {
  int i = blockIdx.x * blockDim.x + threadIdx.x;
  if (i < Dd * Dd) {
    int n = i / Dd, k = i % Dd;
    Wt[(size_t)n * Dd + k] = f2bf(W[(size_t)k * Dd + n]);
  }
}

// ---------- WMMA GEMM: C[MxD] = A[MxD](bf16) @ Wt^T + bias, various epilogues ----------
// mode 0: out1=qu (+add1=u), out2=qv (+add2=v), layout [B,H,S,DH]
// mode 1: out1=K            layout [B,H,S,DH]
// mode 2: out1=Vt           layout [B,H,DH,S]
// mode 3: out1=kr           layout [H,S,DH]   (A has M=S rows)
// mode 4: outf (f32)        layout [M,D]
__global__ __launch_bounds__(256)
void gemm_kernel(const unsigned short* __restrict__ A,
                 const unsigned short* __restrict__ Wt,
                 const float* __restrict__ bias,
                 const float* __restrict__ add1,
                 const float* __restrict__ add2,
                 unsigned short* __restrict__ out1,
                 unsigned short* __restrict__ out2,
                 float* __restrict__ outf, int mode) {
  int lane = threadIdx.x & 31;
  int wave = threadIdx.x >> 5;
  int row0 = blockIdx.x * 16;
  int n0 = (blockIdx.y * NWAVE + wave) * 16;

  v8f c = vzero8();
  for (int kk = 0; kk < Dd; kk += 32) {
    v16bf a = load_afrag(A, Dd, row0, kk, lane);
    v16bf b = load_bfrag(Wt, Dd, n0, kk, lane);
    c = WMMA_BF16(a, b, c);
  }

  int n = n0 + (lane & 15);
  int h = n >> 6, dh = n & 63;
  float bval = bias[n];
  float a1 = add1 ? add1[n] : 0.f;
  float a2 = add2 ? add2[n] : 0.f;

  for (int k = 0; k < 8; ++k) {
    int g = row0 + k + ((lane >> 4) << 3);
    float val = c[k] + bval;
    if (mode == 4) {
      outf[(size_t)g * Dd + n] = val;
    } else if (mode == 3) {
      out1[((size_t)h * Ss + g) * DHh + dh] = f2bf(val);
    } else if (mode == 2) {
      int b_ = g / Ss, s_ = g % Ss;
      out1[(((size_t)(b_ * Hh + h)) * DHh + dh) * Ss + s_] = f2bf(val);
    } else {
      int b_ = g / Ss, s_ = g % Ss;
      size_t idx = (((size_t)(b_ * Hh + h)) * Ss + s_) * DHh + dh;
      out1[idx] = f2bf(val + a1);
      if (out2) out2[idx] = f2bf(val + a2);
    }
  }
}

// ---------- fused relative attention (flash style, split-K softmax over 8 waves) ----------
__global__ __launch_bounds__(256)
void attn_kernel(const unsigned short* __restrict__ quG,
                 const unsigned short* __restrict__ qvG,
                 const unsigned short* __restrict__ KG,
                 const unsigned short* __restrict__ VtG,
                 const unsigned short* __restrict__ krG,
                 unsigned short* __restrict__ outG) {
  extern __shared__ char smem[];
  float* Tq        = (float*)smem;                          // 17*Ss f32
  float* qvrow     = Tq + 17 * Ss;                          // 64
  float* scoreAll  = qvrow + 64;                            // NWAVE*16*32 f32
  unsigned short* probAll = (unsigned short*)(scoreAll + NWAVE * 512); // NWAVE*16*32 bf16
  float* corrBuf   = (float*)(probAll + NWAVE * 512);       // NWAVE*16
  float* statsM    = corrBuf + NWAVE * 16;                  // NWAVE*16
  float* statsL    = statsM + NWAVE * 16;                   // NWAVE*16
  float* Mstar     = statsL + NWAVE * 16;                   // 16
  float* Lstar     = Mstar + 16;                            // 16
  float* factorB   = Lstar + 16;                            // NWAVE*16
  float* Opar      = factorB + NWAVE * 16;                  // NWAVE*16*64 f32

  const int tid = threadIdx.x;
  const int lane = tid & 31, wave = tid >> 5;
  const int nblkS = Ss / 16;
  const int i0 = (blockIdx.x % nblkS) * 16;
  const int bh = blockIdx.x / nblkS;
  const int h = bh % Hh, bIdx = bh / Hh;

  const unsigned short* qu = quG + (size_t)bh * Ss * DHh;
  const unsigned short* qv = qvG + (size_t)bh * Ss * DHh;
  const unsigned short* Km = KG + (size_t)bh * Ss * DHh;
  const unsigned short* Vt = VtG + (size_t)bh * DHh * Ss;
  const unsigned short* kr = krG + (size_t)h * Ss * DHh;

  // ---- phase 1: Tq[r][m] = (q+v)[i0+r] . kr[m] for r=0..16, all m (WMMA + 1 VALU row) ----
  {
    v16bf aq0 = load_afrag(qv, DHh, i0, 0, lane);
    v16bf aq1 = load_afrag(qv, DHh, i0, 32, lane);
    for (int t = 0; t < 16; ++t) {
      int m0 = (wave + t * NWAVE) * 16;
      v16bf b0 = load_bfrag(kr, DHh, m0, 0, lane);
      v16bf b1 = load_bfrag(kr, DHh, m0, 32, lane);
      v8f c = vzero8();
      c = WMMA_BF16(aq0, b0, c);
      c = WMMA_BF16(aq1, b1, c);
      int col = m0 + (lane & 15);
      for (int k = 0; k < 8; ++k)
        Tq[(k + ((lane >> 4) << 3)) * Ss + col] = c[k];
    }
  }
  bool hasRow16 = (i0 + 16) < Ss;
  if (tid < DHh) qvrow[tid] = hasRow16 ? bf2f(qv[(size_t)(i0 + 16) * DHh + tid]) : 0.f;
  __syncthreads();
  for (int mm = 0; mm < 8; ++mm) {
    int m = tid + 256 * mm;
    float acc = 0.f;
    for (int d = 0; d < DHh; ++d) acc += qvrow[d] * bf2f(kr[(size_t)m * DHh + d]);
    Tq[16 * Ss + m] = acc;
  }
  __syncthreads();

  // ---- phase 2: flash attention; wave w owns keys [w*256, w*256+256) ----
  v16bf au0 = load_afrag(qu, DHh, i0, 0, lane);
  v16bf au1 = load_afrag(qu, DHh, i0, 32, lane);
  float* scoreW = scoreAll + wave * 512;
  unsigned short* probW = probAll + wave * 512;
  v8f o0 = vzero8(), o1 = vzero8(), o2 = vzero8(), o3 = vzero8();
  float mrow = -3.0e38f, lrow = 0.f;   // lanes 0..15 own row stats
  const float scale = 0.125f;          // 1/sqrt(DH)
  const int jbase = wave * (Ss / NWAVE);

  for (int it = 0; it < (Ss / NWAVE) / 32; ++it) {
    int j0 = jbase + it * 32;
    v8f c0 = vzero8(), c1 = vzero8();
    {
      v16bf bb;
      bb = load_bfrag(Km, DHh, j0, 0, lane);       c0 = WMMA_BF16(au0, bb, c0);
      bb = load_bfrag(Km, DHh, j0, 32, lane);      c0 = WMMA_BF16(au1, bb, c0);
      bb = load_bfrag(Km, DHh, j0 + 16, 0, lane);  c1 = WMMA_BF16(au0, bb, c1);
      bb = load_bfrag(Km, DHh, j0 + 16, 32, lane); c1 = WMMA_BF16(au1, bb, c1);
    }
    // scores + relative-shift BD gather (exact view_as trick indexing)
    int nloc = lane & 15;
    for (int k = 0; k < 8; ++k) {
      int rr = k + ((lane >> 4) << 3);
      int i = i0 + rr;
      {
        int j = j0 + nloc;
        int f = (i + 1) * Ss + j;
        int r = f / (Ss + 1), cc = f % (Ss + 1);
        float bd = (cc == 0) ? 0.f : Tq[(r - i0) * Ss + (cc - 1)];
        scoreW[rr * 32 + nloc] = (c0[k] + bd) * scale;
      }
      {
        int j = j0 + 16 + nloc;
        int f = (i + 1) * Ss + j;
        int r = f / (Ss + 1), cc = f % (Ss + 1);
        float bd = (cc == 0) ? 0.f : Tq[(r - i0) * Ss + (cc - 1)];
        scoreW[rr * 32 + 16 + nloc] = (c1[k] + bd) * scale;
      }
    }
    __syncthreads();
    if (lane < 16) {  // lane owns query row `lane`
      float mt = -3.0e38f;
      for (int q = 0; q < 32; ++q) mt = fmaxf(mt, scoreW[lane * 32 + q]);
      float mn = fmaxf(mrow, mt);
      float corr = __expf(mrow - mn);
      float ssum = 0.f;
      for (int q = 0; q < 32; ++q) {
        float p = __expf(scoreW[lane * 32 + q] - mn);
        probW[lane * 32 + q] = f2bf(p);
        ssum += p;
      }
      lrow = lrow * corr + ssum;
      mrow = mn;
      corrBuf[wave * 16 + lane] = corr;
    }
    __syncthreads();
    // rescale accumulators, then P @ V
    float cf[8];
    for (int k = 0; k < 8; ++k) cf[k] = corrBuf[wave * 16 + k + ((lane >> 4) << 3)];
    for (int k = 0; k < 8; ++k) { o0[k] *= cf[k]; o1[k] *= cf[k]; o2[k] *= cf[k]; o3[k] *= cf[k]; }
    v16bf pa = load_afrag(probW, 32, 0, 0, lane);
    v16bf vb;
    vb = load_bfrag(Vt, Ss, 0, j0, lane);  o0 = WMMA_BF16(pa, vb, o0);
    vb = load_bfrag(Vt, Ss, 16, j0, lane); o1 = WMMA_BF16(pa, vb, o1);
    vb = load_bfrag(Vt, Ss, 32, j0, lane); o2 = WMMA_BF16(pa, vb, o2);
    vb = load_bfrag(Vt, Ss, 48, j0, lane); o3 = WMMA_BF16(pa, vb, o3);
  }

  // ---- phase 3: deterministic cross-wave softmax merge ----
  if (lane < 16) { statsM[wave * 16 + lane] = mrow; statsL[wave * 16 + lane] = lrow; }
  __syncthreads();
  if (tid < 16) {
    float mg = -3.0e38f;
    for (int w = 0; w < NWAVE; ++w) mg = fmaxf(mg, statsM[w * 16 + tid]);
    float lg = 0.f;
    for (int w = 0; w < NWAVE; ++w) lg += statsL[w * 16 + tid] * __expf(statsM[w * 16 + tid] - mg);
    Mstar[tid] = mg; Lstar[tid] = lg;
  }
  __syncthreads();
  if (lane < 16) factorB[wave * 16 + lane] = __expf(mrow - Mstar[lane]);
  __syncthreads();
  {
    float ff[8];
    for (int k = 0; k < 8; ++k) ff[k] = factorB[wave * 16 + k + ((lane >> 4) << 3)];
    int dhl = lane & 15;
    float* Ow = Opar + wave * 16 * DHh;
    for (int k = 0; k < 8; ++k) {
      int rr = k + ((lane >> 4) << 3);
      Ow[rr * DHh + dhl] = o0[k] * ff[k];
      Ow[rr * DHh + 16 + dhl] = o1[k] * ff[k];
      Ow[rr * DHh + 32 + dhl] = o2[k] * ff[k];
      Ow[rr * DHh + 48 + dhl] = o3[k] * ff[k];
    }
  }
  __syncthreads();
  for (int e = tid; e < 16 * DHh; e += 256) {
    float acc = 0.f;
    for (int w = 0; w < NWAVE; ++w) acc += Opar[w * 16 * DHh + e];
    int rr = e / DHh, dh = e % DHh;
    float val = acc / Lstar[rr];
    outG[((size_t)bIdx * Ss + (i0 + rr)) * Dd + h * DHh + dh] = f2bf(val);
  }
}

// ---------- host launch ----------
extern "C" void kernel_launch(void* const* d_in, const int* in_sizes, int n_in,
                              void* d_out, int out_size, void* d_ws, size_t ws_size,
                              hipStream_t stream) {
  (void)in_sizes; (void)n_in; (void)out_size; (void)ws_size;
  const float* x    = (const float*)d_in[0];
  const float* pos  = (const float*)d_in[1];
  const float* Wq   = (const float*)d_in[2];  const float* bq  = (const float*)d_in[3];
  const float* Wk   = (const float*)d_in[4];  const float* bk  = (const float*)d_in[5];
  const float* Wv   = (const float*)d_in[6];  const float* bv  = (const float*)d_in[7];
  const float* Wo   = (const float*)d_in[8];  const float* bo  = (const float*)d_in[9];
  const float* Wkr  = (const float*)d_in[10]; const float* bkr = (const float*)d_in[11];
  const float* u    = (const float*)d_in[12]; const float* v   = (const float*)d_in[13];
  float* out = (float*)d_out;

  char* ws = (char*)d_ws;
  size_t off = 0;
  auto alloc = [&](size_t bytes) { size_t o = off; off += (bytes + 255) & ~(size_t)255; return o; };
  const size_t nBSD = (size_t)Bb * Ss * Dd;      // 4,194,304
  const size_t nSD  = (size_t)Ss * Dd;           // 2,097,152
  const size_t nDD  = (size_t)Dd * Dd;           // 1,048,576

  unsigned short* xb   = (unsigned short*)(ws + alloc(nBSD * 2));
  unsigned short* pb   = (unsigned short*)(ws + alloc(nSD * 2));
  unsigned short* Wqt  = (unsigned short*)(ws + alloc(nDD * 2));
  unsigned short* Wkt  = (unsigned short*)(ws + alloc(nDD * 2));
  unsigned short* Wvt  = (unsigned short*)(ws + alloc(nDD * 2));
  unsigned short* Wot  = (unsigned short*)(ws + alloc(nDD * 2));
  unsigned short* Wkrt = (unsigned short*)(ws + alloc(nDD * 2));
  unsigned short* quB  = (unsigned short*)(ws + alloc(nBSD * 2));
  unsigned short* qvB  = (unsigned short*)(ws + alloc(nBSD * 2));
  unsigned short* Kb   = (unsigned short*)(ws + alloc(nBSD * 2));
  unsigned short* VtB  = (unsigned short*)(ws + alloc(nBSD * 2));
  unsigned short* krB  = (unsigned short*)(ws + alloc(nSD * 2));
  unsigned short* attO = (unsigned short*)(ws + alloc(nBSD * 2));

  // conversions
  cvt_bf16_kernel<<<(int)((nBSD + 255) / 256), 256, 0, stream>>>(x, xb, (int)nBSD);
  cvt_bf16_kernel<<<(int)((nSD + 255) / 256), 256, 0, stream>>>(pos, pb, (int)nSD);
  transpose_cvt_kernel<<<(int)((nDD + 255) / 256), 256, 0, stream>>>(Wq, Wqt);
  transpose_cvt_kernel<<<(int)((nDD + 255) / 256), 256, 0, stream>>>(Wk, Wkt);
  transpose_cvt_kernel<<<(int)((nDD + 255) / 256), 256, 0, stream>>>(Wv, Wvt);
  transpose_cvt_kernel<<<(int)((nDD + 255) / 256), 256, 0, stream>>>(Wo, Wot);
  transpose_cvt_kernel<<<(int)((nDD + 255) / 256), 256, 0, stream>>>(Wkr, Wkrt);

  // projections (M = B*S rows; kr: M = S rows)
  dim3 gP(Bb * Ss / 16, Dd / (16 * NWAVE));
  gemm_kernel<<<gP, 256, 0, stream>>>(xb, Wqt, bq, u, v, quB, qvB, nullptr, 0);
  gemm_kernel<<<gP, 256, 0, stream>>>(xb, Wkt, bk, nullptr, nullptr, Kb, nullptr, nullptr, 1);
  gemm_kernel<<<gP, 256, 0, stream>>>(xb, Wvt, bv, nullptr, nullptr, VtB, nullptr, nullptr, 2);
  dim3 gR(Ss / 16, Dd / (16 * NWAVE));
  gemm_kernel<<<gR, 256, 0, stream>>>(pb, Wkrt, bkr, nullptr, nullptr, krB, nullptr, nullptr, 3);

  // attention
  const int smemBytes = 4 * (17 * Ss + 64 + NWAVE * 512)   // Tq, qvrow, scores
                      + 2 * (NWAVE * 512)                  // probs (bf16)
                      + 4 * (NWAVE * 16 * 3 + 32 + NWAVE * 16 + NWAVE * 16 * DHh);
  (void)hipFuncSetAttribute((const void*)attn_kernel,
                            hipFuncAttributeMaxDynamicSharedMemorySize, smemBytes);
  attn_kernel<<<Bb * Hh * (Ss / 16), 256, smemBytes, stream>>>(quB, qvB, Kb, VtB, krB, attO);

  // output projection -> f32
  gemm_kernel<<<gP, 256, 0, stream>>>(attO, Wot, bo, nullptr, nullptr, nullptr, nullptr, out, 4);
}
